// Encoder_1314259993027
// MI455X (gfx1250) — matrix operand order repaired and checked by reading, not compiled, not run on hardware
//
#include <hip/hip_runtime.h>
#include <cstdint>
#include <cstddef>

// Problem constants (match reference)
#define B_   16
#define S_   512
#define D_   768
#define H_   12
#define DN_  64
#define F_   3072
#define L_   6
#define NT_  (B_ * S_)   // 8192 tokens

typedef unsigned short ushort_t;
typedef __attribute__((ext_vector_type(16))) __bf16 bf16x16;
typedef __attribute__((ext_vector_type(8)))  float  f32x8;
typedef __attribute__((ext_vector_type(4)))  unsigned int uint32x4;
typedef __attribute__((ext_vector_type(8)))  int   int32x8;
typedef __attribute__((ext_vector_type(4)))  int   int32x4;

union FragB16 { bf16x16 v; unsigned int u[8]; };
union FragC   { f32x8   v; float        f[8]; };

__device__ __forceinline__ ushort_t f2bf(float x) {
  union { float f; unsigned int u; } c; c.f = x;
  unsigned int r = c.u + 0x7FFFu + ((c.u >> 16) & 1u);  // round-to-nearest-even
  return (ushort_t)(r >> 16);
}

__device__ __forceinline__ f32x8 wmma_bf16(bf16x16 a, bf16x16 b, f32x8 c) {
  // (neg_a, A, neg_b, B, c_mod, C, reuse_a, reuse_b)
  return __builtin_amdgcn_wmma_f32_16x16x32_bf16(false, a, false, b, (short)0, c,
                                                 false, false);
}

// ---- TDM: async 2D tile load (rows x 32-elem bf16 lines) into LDS ---------
// Descriptor per CDNA5 ISA 8.3/8.4. 2D tensor -> groups 2/3 zero.
// This toolchain (clang-23 / therock-10.0) uses the 6-arg builtin:
// (uint32x4 g0, int32x8 g1, int32x4 g2, int32x4 g3, int32x8 gx, i32 cpol)
__device__ __forceinline__ void tdm_load_tile_bf16(
    unsigned ldsOff, const ushort_t* gptr, int lda /*elements*/,
    int tileRows, int tileCols, int tensorRows) {
  unsigned long long ga = (unsigned long long)(size_t)gptr;
  uint32x4 g0;
  g0[0] = 1u;                                        // count=1, user descriptor
  g0[1] = ldsOff;                                    // LDS byte address
  g0[2] = (unsigned)(ga & 0xFFFFFFFFu);              // global_addr[31:0]
  g0[3] = (unsigned)((ga >> 32) & 0x01FFFFFFu)       // global_addr[56:32]
          | 0x80000000u;                             // type=2 ("image")
  int32x8 g1;
  g1[0] = 0x00010000;                                // data_size=1 (2 bytes)
  g1[1] = (int)(((unsigned)lda & 0xFFFFu) << 16);    // tensor_dim0[15:0]
  g1[2] = (int)((((unsigned)lda >> 16) & 0xFFFFu)    // tensor_dim0[31:16]
          | (((unsigned)tensorRows & 0xFFFFu) << 16));   // tensor_dim1[15:0]
  g1[3] = (int)((((unsigned)tensorRows >> 16) & 0xFFFFu)
          | (((unsigned)tileCols & 0xFFFFu) << 16)); // tile_dim0
  g1[4] = (int)((unsigned)tileRows & 0xFFFFu);       // tile_dim1 (tile_dim2=0)
  g1[5] = lda;                                       // tensor_dim0_stride[31:0]
  g1[6] = 0;                                         // stride hi, dim1_stride lo
  g1[7] = 0;
  int32x4 gz; gz[0] = 0; gz[1] = 0; gz[2] = 0; gz[3] = 0;
  int32x8 gz8;
#pragma unroll
  for (int i = 0; i < 8; ++i) gz8[i] = 0;
  __builtin_amdgcn_tensor_load_to_lds(g0, g1, gz, gz, gz8, 0 /*cpol*/);
}

// ---- fragment loaders (wave32; contiguous k = 16*(lane/16) + s, same order in A and B) ----------------
// A: 16x32 bf16 tile, row-major in LDS with row stride ldr (elements).
// lanes 0-15: M=lane, K 0..15; lanes 16-31: M=lane-16, K 16..31
__device__ __forceinline__ bf16x16 loadA(const ushort_t* lds, int ldr) {
  int lane = threadIdx.x & 31;
  int m = lane & 15;
  int khalf = (lane >> 4) << 4;
  FragB16 a;
#pragma unroll
  for (int vv = 0; vv < 8; ++vv) {
    int k0 = (vv << 1) + khalf;  // even
    a.u[vv] = *(const unsigned int*)(lds + m * ldr + k0);
  }
  return a.v;
}

// B: 32x16 bf16 tile (KxN), source LDS row-major [K][N] with row stride ldr.
// lanes 0-15: K 0..15 ; lanes 16-31: K 16..31 ; N = lane&15
__device__ __forceinline__ bf16x16 loadB(const ushort_t* lds, int ldr) {
  int lane = threadIdx.x & 31;
  int n = lane & 15;
  int kb = (lane >> 4) << 4;
  FragB16 b;
#pragma unroll
  for (int vv = 0; vv < 8; ++vv) {
    int k0 = kb + (vv << 1);
    unsigned int lo = lds[(k0    ) * ldr + n];
    unsigned int hi = lds[(k0 + 1) * ldr + n];
    b.u[vv] = lo | (hi << 16);
  }
  return b.v;
}

// B from a transposed source: B[k][n] = src[n][k] (src row-major 16 x >=32,
// row stride ldr). Pairs along k are contiguous -> dword loads.
__device__ __forceinline__ bf16x16 loadBt(const ushort_t* lds, int ldr) {
  int lane = threadIdx.x & 31;
  int n = lane & 15;
  int kb = (lane >> 4) << 4;
  FragB16 b;
#pragma unroll
  for (int vv = 0; vv < 8; ++vv) {
    int k0 = kb + (vv << 1);
    b.u[vv] = *(const unsigned int*)(lds + n * ldr + k0);
  }
  return b.v;
}

// ---- kernel 1: embedding + positional encoding -----------------------------
__global__ void embed_pe_kernel(const int* __restrict__ x,
                                const float* __restrict__ emb,
                                float* __restrict__ h, ushort_t* __restrict__ hb) {
  int row = blockIdx.x;          // token index 0..NT-1
  int s   = row & (S_ - 1);
  int tok = x[row];
  const float LOG1E4 = 9.210340371976184f;  // ln(10000)
  for (int d = threadIdx.x; d < D_; d += blockDim.x) {
    int i = d >> 1;
    float pe;
    if ((d & 1) == 0) {
      float freq = __expf(-(2.0f * (float)i / (float)D_) * LOG1E4);
      pe = __sinf((float)s * freq);
    } else {
      float freq = __expf(-(2.0f * (float)(i + 1) / (float)D_) * LOG1E4);
      pe = __cosf((float)s * freq);
    }
    float v = emb[(size_t)tok * D_ + d] + pe;
    h [(size_t)row * D_ + d] = v;
    hb[(size_t)row * D_ + d] = f2bf(v);
  }
}

// ---- kernel 2: tiled bf16 WMMA GEMM ---------------------------------------
// C[M,N] = A[M,K](bf16) * W[K,N](f32, converted in staging) + bias
// A tile is DMA'd to LDS by the Tensor Data Mover (TENSORcnt); B tile is
// staged manually with on-the-fly f32->bf16 conversion.
// BMODE 0: W row-major [K, Ncols]; BMODE 1: head-blocked Wq-style [H, D, DN]
// OMODE 0: f32 out (ld = ldo); OMODE 1: bf16 QKV -> [B,H,S,DN]; OMODE 2: relu->bf16
template <int BMODE, int OMODE>
__global__ void gemm_kernel(const ushort_t* __restrict__ A, int lda,
                            const float* __restrict__ W,
                            const float* __restrict__ bias,
                            void* __restrict__ Out, int ldo, int K, int Ncols) {
  __shared__ __align__(16) ushort_t sA[128 * 32];
  __shared__ __align__(16) ushort_t sB[32 * 128];
  int tid  = threadIdx.x;
  int lane = tid & 31, wave = tid >> 5;
  int wm = wave >> 2, wn = wave & 3;            // 2x4 wave grid -> 64x32 per wave
  int rowBase = blockIdx.y * 128;
  int colBase = blockIdx.x * 128;
  unsigned sAoff = (unsigned)(size_t)(&sA[0]);  // LDS byte offset (addr[31:0])

  FragC c[4][2];
#pragma unroll
  for (int i = 0; i < 4; ++i)
#pragma unroll
    for (int j = 0; j < 2; ++j)
#pragma unroll
      for (int r = 0; r < 8; ++r) c[i][j].f[r] = 0.0f;

  int nk = K >> 5;
  for (int kk = 0; kk < nk; ++kk) {
    // A tile 128x32 bf16 via TDM (wave 0 issues; barrier publishes)
    if (tid < 32) {
      tdm_load_tile_bf16(sAoff, A + (size_t)rowBase * lda + (kk << 5), lda,
                         128, 32, NT_ - rowBase);
    }
    // stage B tile 32x128, f32 -> bf16 conversion on the fly
#pragma unroll
    for (int it = 0; it < 16; ++it) {
      int idx = tid + it * 256;                 // 4096 elems
      int kr = idx >> 7;
      int n  = idx & 127;
      int gk = (kk << 5) + kr;
      int gc = colBase + n;
      float wv;
      if (BMODE == 0)
        wv = W[(size_t)gk * Ncols + gc];
      else
        wv = W[(size_t)(gc >> 6) * (D_ * DN_) + (size_t)gk * DN_ + (gc & 63)];
      sB[kr * 128 + n] = f2bf(wv);
    }
    if (kk + 1 < nk && BMODE == 0) {            // speculative prefetch of next W tile
      __builtin_prefetch(W + (size_t)((kk + 1) << 5) * Ncols + colBase + (tid >> 1), 0, 0);
    }
    if (tid < 32) {
      __builtin_amdgcn_s_wait_tensorcnt(0);     // A tile resident in LDS
    }
    __syncthreads();

    bf16x16 afr[4];
#pragma unroll
    for (int mt = 0; mt < 4; ++mt)
      afr[mt] = loadA(sA + (wm * 64 + mt * 16) * 32, 32);
    bf16x16 bfr[2];
#pragma unroll
    for (int nt = 0; nt < 2; ++nt)
      bfr[nt] = loadB(sB + wn * 32 + nt * 16, 128);
#pragma unroll
    for (int mt = 0; mt < 4; ++mt)
#pragma unroll
      for (int nt = 0; nt < 2; ++nt)
        c[mt][nt].v = wmma_bf16(afr[mt], bfr[nt], c[mt][nt].v);
    __syncthreads();
  }

  // epilogue: C layout M = r + 8*(lane>=16), N = lane&15
  int mrow = (lane >> 4) << 3;
  int ncol = lane & 15;
#pragma unroll
  for (int mt = 0; mt < 4; ++mt) {
#pragma unroll
    for (int nt = 0; nt < 2; ++nt) {
#pragma unroll
      for (int r = 0; r < 8; ++r) {
        int row = rowBase + wm * 64 + mt * 16 + mrow + r;
        int col = colBase + wn * 32 + nt * 16 + ncol;
        float v = c[mt][nt].f[r] + bias[col];
        if (OMODE == 0) {
          ((float*)Out)[(size_t)row * ldo + col] = v;
        } else if (OMODE == 2) {
          ((ushort_t*)Out)[(size_t)row * ldo + col] = f2bf(v > 0.0f ? v : 0.0f);
        } else {  // QKV: [B,H,S,DN]
          int b = row >> 9, s = row & 511;
          int hh = col >> 6, e = col & 63;
          ((ushort_t*)Out)[(((size_t)(b * H_ + hh)) * S_ + s) * DN_ + e] = f2bf(v);
        }
      }
    }
  }
}

// ---- kernel 3: attention, one wave per (b, h, 16-row query tile) ----------
__global__ void attn_kernel(const ushort_t* __restrict__ q,
                            const ushort_t* __restrict__ k,
                            const ushort_t* __restrict__ v,
                            const int* __restrict__ mask,
                            ushort_t* __restrict__ o) {
  __shared__ __align__(16) ushort_t sQ[16 * 64];
  __shared__ __align__(16) ushort_t sK[16 * 64];
  __shared__ __align__(16) ushort_t sV[32 * 64];
  __shared__ __align__(16) float    sS[16 * 512];
  __shared__ __align__(16) ushort_t sP[16 * 512];

  int bidx = blockIdx.x;            // (b*H + h)*32 + mtile
  int mt   = bidx & 31;
  int bh   = bidx >> 5;
  int hh   = bh % H_;
  int b    = bh / H_;
  int srow = mt << 4;
  int lane = threadIdx.x;

  const ushort_t* qb = q + (size_t)bh * S_ * DN_;
  const ushort_t* kb = k + (size_t)bh * S_ * DN_;
  const ushort_t* vb = v + (size_t)bh * S_ * DN_;

  { // load Q tile 16x64 (128 x uint4)
    const uint4* src = (const uint4*)(qb + srow * DN_);
    uint4* dst = (uint4*)sQ;
#pragma unroll
    for (int i = 0; i < 4; ++i) dst[lane + 32 * i] = src[lane + 32 * i];
  }
  __syncthreads();

  const float scale = 0.125f;       // 1/sqrt(64)
  for (int kt = 0; kt < 32; ++kt) {
    const uint4* src = (const uint4*)(kb + (kt << 4) * DN_);
    uint4* dst = (uint4*)sK;
#pragma unroll
    for (int i = 0; i < 4; ++i) dst[lane + 32 * i] = src[lane + 32 * i];
    __syncthreads();

    FragC cc;
#pragma unroll
    for (int r = 0; r < 8; ++r) cc.f[r] = 0.0f;
    cc.v = wmma_bf16(loadA(sQ,      64), loadBt(sK,      64), cc.v);  // K=0..31
    cc.v = wmma_bf16(loadA(sQ + 32, 64), loadBt(sK + 32, 64), cc.v);  // K=32..63

    int mrow = (lane >> 4) << 3, n = lane & 15;
#pragma unroll
    for (int r = 0; r < 8; ++r) {
      int m = mrow + r;
      int colg = (kt << 4) + n;
      float bias = (float)mask[((size_t)b * S_ + (srow + m)) * S_ + colg] * -1e9f;
      sS[m * 512 + colg] = cc.f[r] * scale + bias;
    }
    __syncthreads();
  }

  // f32 softmax over 16 rows of 512, probs -> bf16 in sP
  for (int m = 0; m < 16; ++m) {
    float mx = -3.0e38f;
    for (int ci = lane; ci < 512; ci += 32) mx = fmaxf(mx, sS[m * 512 + ci]);
#pragma unroll
    for (int off = 16; off > 0; off >>= 1) mx = fmaxf(mx, __shfl_xor(mx, off, 32));
    float sum = 0.0f;
    for (int ci = lane; ci < 512; ci += 32) {
      float e = __expf(sS[m * 512 + ci] - mx);
      sS[m * 512 + ci] = e;
      sum += e;
    }
#pragma unroll
    for (int off = 16; off > 0; off >>= 1) sum += __shfl_xor(sum, off, 32);
    float inv = 1.0f / sum;
    for (int ci = lane; ci < 512; ci += 32)
      sP[m * 512 + ci] = f2bf(sS[m * 512 + ci] * inv);
  }
  __syncthreads();

  // O = P[16,512] x V[512,64]
  FragC oacc[4];
#pragma unroll
  for (int nt = 0; nt < 4; ++nt)
#pragma unroll
    for (int r = 0; r < 8; ++r) oacc[nt].f[r] = 0.0f;

  for (int kc = 0; kc < 16; ++kc) {
    const uint4* src = (const uint4*)(vb + (kc << 5) * DN_);
    uint4* dst = (uint4*)sV;
#pragma unroll
    for (int i = 0; i < 8; ++i) dst[lane + 32 * i] = src[lane + 32 * i];
    __syncthreads();
    bf16x16 ap = loadA(sP + (kc << 5), 512);
#pragma unroll
    for (int nt = 0; nt < 4; ++nt)
      oacc[nt].v = wmma_bf16(ap, loadB(sV + nt * 16, 64), oacc[nt].v);
    __syncthreads();
  }

  int mrow = (lane >> 4) << 3, n = lane & 15;
#pragma unroll
  for (int nt = 0; nt < 4; ++nt)
#pragma unroll
    for (int r = 0; r < 8; ++r) {
      int sg  = srow + mrow + r;
      int col = hh * 64 + nt * 16 + n;
      o[((size_t)(b * S_ + sg)) * D_ + col] = f2bf(oacc[nt].f[r]);
    }
}

// ---- kernel 4: residual + LayerNorm (block per row) -----------------------
__global__ void resid_ln_kernel(float* __restrict__ h, const float* __restrict__ add,
                                const float* __restrict__ g, const float* __restrict__ be,
                                ushort_t* __restrict__ hb) {
  __shared__ float red[256];
  int row = blockIdx.x;
  int tid = threadIdx.x;
  float vals[3];
  float s = 0.0f;
#pragma unroll
  for (int i = 0; i < 3; ++i) {
    int d = tid + i * 256;
    float z = h[(size_t)row * D_ + d] + add[(size_t)row * D_ + d];
    vals[i] = z; s += z;
  }
  red[tid] = s; __syncthreads();
  for (int off = 128; off > 0; off >>= 1) {
    if (tid < off) red[tid] += red[tid + off];
    __syncthreads();
  }
  float mean = red[0] * (1.0f / D_);
  __syncthreads();
  float vsum = 0.0f;
#pragma unroll
  for (int i = 0; i < 3; ++i) { float dd = vals[i] - mean; vsum += dd * dd; }
  red[tid] = vsum; __syncthreads();
  for (int off = 128; off > 0; off >>= 1) {
    if (tid < off) red[tid] += red[tid + off];
    __syncthreads();
  }
  float rstd = rsqrtf(red[0] * (1.0f / D_) + 1e-5f);
#pragma unroll
  for (int i = 0; i < 3; ++i) {
    int d = tid + i * 256;
    float out = (vals[i] - mean) * rstd * g[d] + be[d];
    h [(size_t)row * D_ + d] = out;
    hb[(size_t)row * D_ + d] = f2bf(out);
  }
}

// ---- host launcher --------------------------------------------------------
extern "C" void kernel_launch(void* const* d_in, const int* in_sizes, int n_in,
                              void* d_out, int out_size, void* d_ws, size_t ws_size,
                              hipStream_t stream) {
  (void)in_sizes; (void)n_in; (void)out_size; (void)ws_size;
  const int*   x    = (const int*)d_in[0];
  const int*   mask = (const int*)d_in[1];
  const float* emb  = (const float*)d_in[2];
  const float* Wq   = (const float*)d_in[3];
  const float* bq   = (const float*)d_in[4];
  const float* Wk   = (const float*)d_in[5];
  const float* bk   = (const float*)d_in[6];
  const float* Wv   = (const float*)d_in[7];
  const float* bv   = (const float*)d_in[8];
  const float* Wo   = (const float*)d_in[9];
  const float* bo   = (const float*)d_in[10];
  const float* W1   = (const float*)d_in[11];
  const float* b1   = (const float*)d_in[12];
  const float* W2   = (const float*)d_in[13];
  const float* b2   = (const float*)d_in[14];
  const float* g1   = (const float*)d_in[15];
  const float* be1  = (const float*)d_in[16];
  const float* g2   = (const float*)d_in[17];
  const float* be2  = (const float*)d_in[18];

  // workspace layout (~164 MB)
  char* ws = (char*)d_ws;
  float*    h   = (float*)ws;      ws += (size_t)NT_ * D_ * sizeof(float);
  ushort_t* hb  = (ushort_t*)ws;   ws += (size_t)NT_ * D_ * sizeof(ushort_t);
  ushort_t* qb  = (ushort_t*)ws;   ws += (size_t)NT_ * D_ * sizeof(ushort_t);
  ushort_t* kb  = (ushort_t*)ws;   ws += (size_t)NT_ * D_ * sizeof(ushort_t);
  ushort_t* vb  = (ushort_t*)ws;   ws += (size_t)NT_ * D_ * sizeof(ushort_t);
  ushort_t* ob  = (ushort_t*)ws;   ws += (size_t)NT_ * D_ * sizeof(ushort_t);
  ushort_t* ffh = (ushort_t*)ws;   ws += (size_t)NT_ * F_ * sizeof(ushort_t);
  float*    tmp = (float*)ws;      ws += (size_t)NT_ * D_ * sizeof(float);

  embed_pe_kernel<<<NT_, 256, 0, stream>>>(x, emb, h, hb);

  dim3 blk(256);
  dim3 gD (D_ / 128, NT_ / 128);   // N=768
  dim3 gF (F_ / 128, NT_ / 128);   // N=3072

  for (int l = 0; l < L_; ++l) {
    const float* Wql = Wq + (size_t)l * H_ * D_ * DN_;
    const float* Wkl = Wk + (size_t)l * H_ * D_ * DN_;
    const float* Wvl = Wv + (size_t)l * H_ * D_ * DN_;
    const float* bql = bq + (size_t)l * H_ * DN_;
    const float* bkl = bk + (size_t)l * H_ * DN_;
    const float* bvl = bv + (size_t)l * H_ * DN_;
    const float* Wol = Wo + (size_t)l * D_ * D_;
    const float* bol = bo + (size_t)l * D_;
    const float* W1l = W1 + (size_t)l * D_ * F_;
    const float* b1l = b1 + (size_t)l * F_;
    const float* W2l = W2 + (size_t)l * F_ * D_;
    const float* b2l = b2 + (size_t)l * D_;

    gemm_kernel<1, 1><<<gD, blk, 0, stream>>>(hb, D_, Wql, bql, qb, 0, D_, D_);
    gemm_kernel<1, 1><<<gD, blk, 0, stream>>>(hb, D_, Wkl, bkl, kb, 0, D_, D_);
    gemm_kernel<1, 1><<<gD, blk, 0, stream>>>(hb, D_, Wvl, bvl, vb, 0, D_, D_);

    attn_kernel<<<B_ * H_ * (S_ / 16), 32, 0, stream>>>(qb, kb, vb, mask, ob);

    gemm_kernel<0, 0><<<gD, blk, 0, stream>>>(ob, D_, Wol, bol, tmp, D_, D_, D_);
    resid_ln_kernel<<<NT_, 256, 0, stream>>>(h, tmp, g1 + (size_t)l * D_,
                                             be1 + (size_t)l * D_, hb);

    gemm_kernel<0, 2><<<gF, blk, 0, stream>>>(hb, D_, W1l, b1l, ffh, F_, D_, F_);
    gemm_kernel<0, 0><<<gD, blk, 0, stream>>>(ffh, F_, W2l, b2l, tmp, D_, F_, D_);
    resid_ln_kernel<<<NT_, 256, 0, stream>>>(h, tmp, g2 + (size_t)l * D_,
                                             be2 + (size_t)l * D_, hb);
  }

  (void)hipMemcpyAsync(d_out, h, (size_t)NT_ * D_ * sizeof(float),
                       hipMemcpyDeviceToDevice, stream);
}
